// MultiHeadedAttentionBlock_4587025072442
// MI455X (gfx1250) — compile-verified, matrix operand order checked by reading, not run on hardware
//
#include <hip/hip_runtime.h>
#include <math.h>
#include <stdint.h>

// ---------------------------------------------------------------------------
// MultiHeadedAttentionBlock for MI455X (gfx1250), bf16 WMMA pipeline.
// B=2, S=2048, D=1024, H=16, head_dim=64.
// ---------------------------------------------------------------------------

typedef __attribute__((ext_vector_type(16))) __bf16 v16bf;
typedef __attribute__((ext_vector_type(8)))  __bf16 v8bf;
typedef __attribute__((ext_vector_type(8)))  float  v8f;
typedef __attribute__((ext_vector_type(4)))  unsigned int u32x4;
typedef __attribute__((ext_vector_type(8)))  int i32x8;
typedef __attribute__((ext_vector_type(4)))  int i32x4;

#define DEV static __device__ __forceinline__

constexpr int BB = 2;
constexpr int SS = 2048;
constexpr int DD = 1024;
constexpr int HH = 16;
constexpr int HD = 64;            // head dim
constexpr int MM = BB * SS;       // 4096 rows

// float -> bf16 round-to-nearest-even
DEV __bf16 f2bf(float f) {
  unsigned u = __builtin_bit_cast(unsigned, f);
  unsigned r = (u + 0x7FFFu + ((u >> 16) & 1u)) >> 16;
  unsigned short s = (unsigned short)r;
  return __builtin_bit_cast(__bf16, s);
}

DEV v16bf mk16(v8bf lo, v8bf hi) {
  v16bf r;
#pragma unroll
  for (int i = 0; i < 8; ++i) { r[i] = lo[i]; r[i + 8] = hi[i]; }
  return r;
}

DEV v8f wmma_bf16(v16bf a, v16bf b, v8f c) {
  return __builtin_amdgcn_wmma_f32_16x16x32_bf16(false, a, false, b,
                                                 (short)0, c, false, false);
}

// ---------------------------------------------------------------------------
// fp32 -> bf16 elementwise convert
// ---------------------------------------------------------------------------
__global__ void cvt_f32_bf16(const float* __restrict__ src,
                             __bf16* __restrict__ dst, int n) {
  int i = blockIdx.x * blockDim.x + threadIdx.x;
  if (i < n) dst[i] = f2bf(src[i]);
}

// ---------------------------------------------------------------------------
// C[m,n] = sum_k A[m,k] * W[n,k]   (x @ W^T), bf16 inputs, f32 accum.
// Wave tile: 32 rows x 64 cols -> 8 accumulators, 8 WMMAs per K-step of 32
// with only 12 b128 loads (A reused x4, B reused x2). Register
// double-buffering: fragments for k+32 are fetched before computing k, so
// waits cover only older loads. Block = 8 waves = 64x256 output tile.
// ---------------------------------------------------------------------------
__global__ __launch_bounds__(256)
void gemm_bf16_wmma(const __bf16* __restrict__ A, const __bf16* __restrict__ W,
                    __bf16* __restrict__ outb, float* __restrict__ outf,
                    const float* __restrict__ bias, int M, int N, int K) {
  const int lane = threadIdx.x & 31;
  const int w    = threadIdx.x >> 5;
  const int half = lane >> 4;
  const int lcol = lane & 15;
  const int row0 = blockIdx.y * 64 + (w >> 2) * 32;
  const int col0 = blockIdx.x * 256 + (w & 3) * 64;
  (void)M;

  const __bf16* ar0 = A + (size_t)(row0 + lcol) * K;
  const __bf16* ar1 = A + (size_t)(row0 + 16 + lcol) * K;
  const __bf16* wr0 = W + (size_t)(col0 + 0 * 16 + lcol) * K;
  const __bf16* wr1 = W + (size_t)(col0 + 1 * 16 + lcol) * K;
  const __bf16* wr2 = W + (size_t)(col0 + 2 * 16 + lcol) * K;
  const __bf16* wr3 = W + (size_t)(col0 + 3 * 16 + lcol) * K;

  auto ldfrag = [&](const __bf16* p, int k) {
    return mk16(*(const v8bf*)(p + k + half * 8),
                *(const v8bf*)(p + k + 16 + half * 8));
  };

  v8f acc[8] = {};

  v16bf a0 = ldfrag(ar0, 0), a1 = ldfrag(ar1, 0);
  v16bf b0 = ldfrag(wr0, 0), b1 = ldfrag(wr1, 0);
  v16bf b2 = ldfrag(wr2, 0), b3 = ldfrag(wr3, 0);

  for (int k = 32; k <= K - 32; k += 32) {
    // prefetch next K-step fragments before consuming current ones
    v16bf na0 = ldfrag(ar0, k), na1 = ldfrag(ar1, k);
    v16bf nb0 = ldfrag(wr0, k), nb1 = ldfrag(wr1, k);
    v16bf nb2 = ldfrag(wr2, k), nb3 = ldfrag(wr3, k);

    acc[0] = wmma_bf16(a0, b0, acc[0]);
    acc[1] = wmma_bf16(a0, b1, acc[1]);
    acc[2] = wmma_bf16(a0, b2, acc[2]);
    acc[3] = wmma_bf16(a0, b3, acc[3]);
    acc[4] = wmma_bf16(a1, b0, acc[4]);
    acc[5] = wmma_bf16(a1, b1, acc[5]);
    acc[6] = wmma_bf16(a1, b2, acc[6]);
    acc[7] = wmma_bf16(a1, b3, acc[7]);

    a0 = na0; a1 = na1; b0 = nb0; b1 = nb1; b2 = nb2; b3 = nb3;
  }
  acc[0] = wmma_bf16(a0, b0, acc[0]);
  acc[1] = wmma_bf16(a0, b1, acc[1]);
  acc[2] = wmma_bf16(a0, b2, acc[2]);
  acc[3] = wmma_bf16(a0, b3, acc[3]);
  acc[4] = wmma_bf16(a1, b0, acc[4]);
  acc[5] = wmma_bf16(a1, b1, acc[5]);
  acc[6] = wmma_bf16(a1, b2, acc[6]);
  acc[7] = wmma_bf16(a1, b3, acc[7]);

#pragma unroll
  for (int rr = 0; rr < 2; ++rr) {
#pragma unroll
    for (int g = 0; g < 4; ++g) {
      int n = col0 + g * 16 + lcol;
      float bv = bias ? bias[n] : 0.0f;
#pragma unroll
      for (int r = 0; r < 8; ++r) {
        int m = row0 + rr * 16 + half * 8 + r;   // C layout: M = 8*half + r
        float v = acc[rr * 4 + g][r] + bv;
        if (outf) outf[(size_t)m * N + n] = v;
        else      outb[(size_t)m * N + n] = f2bf(v);
      }
    }
  }
}

// ---------------------------------------------------------------------------
// Flash-style attention. Block = 8 waves; each wave owns one 16-row q tile
// (block covers 128 q rows of one (b,h)). K/V kv-tiles (32 cols) are staged
// ONCE per block in LDS and shared by all 8 waves (8x less L2 traffic):
//   - K tile (32x64, straight 2D copy) via Tensor Data Mover from wave 0
//   - V tile staged transposed (Vt[d][j]) cooperatively by all 256 threads
// Online softmax per wave; P re-laid out C->A through a per-wave LDS slice.
// ---------------------------------------------------------------------------
__global__ __launch_bounds__(256)
void attn_wmma(const __bf16* __restrict__ Qb, const __bf16* __restrict__ Kb,
               const __bf16* __restrict__ Vb,
               const unsigned char* __restrict__ kvmask,
               __bf16* __restrict__ Ob) {
  __shared__ __bf16 Klds[32 * 64];      // K rows j(32) x d(64), 4 KB
  __shared__ __bf16 Vlds[64 * 32];      // Vt[d][j], 4 KB
  __shared__ __bf16 Plds[8][16 * 32];   // per-wave P tile, 8 KB

  const int tid  = threadIdx.x;
  const int lane = tid & 31;
  const int wv   = tid >> 5;
  const int half = lane >> 4;
  const int lcol = lane & 15;
  const int b  = blockIdx.y / HH;
  const int h  = blockIdx.y % HH;
  const int q0 = blockIdx.x * 128 + wv * 16;
  const size_t base = (size_t)b * SS * DD + (size_t)h * HD;

  // Q A-fragments: rows q0..q0+15, head dim 64 as two K-steps of 32.
  const __bf16* qrow = Qb + base + (size_t)(q0 + lcol) * DD;
  v16bf Qa0 = mk16(*(const v8bf*)(qrow + half * 8),
                   *(const v8bf*)(qrow + 16 + half * 8));
  v16bf Qa1 = mk16(*(const v8bf*)(qrow + 32 + half * 8),
                   *(const v8bf*)(qrow + 48 + half * 8));

  v8f o[4] = {};
  float mrow[8], lrow[8];
#pragma unroll
  for (int r = 0; r < 8; ++r) { mrow[r] = -3.0e38f; lrow[r] = 0.0f; }

  for (int j0 = 0; j0 < SS; j0 += 32) {
    // ---- wave 0: TDM load of K tile (32 rows x 64 bf16, row stride D) ----
    if (wv == 0) {
      const unsigned lds_base = (unsigned)(uintptr_t)(void*)&Klds[0];
      const unsigned long long ga =
          (unsigned long long)(uintptr_t)(Kb + base + (size_t)j0 * DD);
      u32x4 g0 = { 1u,                                     // count=1
                   lds_base,                               // lds_addr
                   (unsigned)(ga & 0xFFFFFFFFu),           // global_addr lo
                   (unsigned)((ga >> 32) & 0x01FFFFFFu) | (2u << 30) }; // +type=2
      i32x8 g1 = { (int)(1u << 16),                        // data_size=2B
                   (int)((DD & 0xFFFF) << 16),             // tensor_dim0 lo16
                   (int)((SS & 0xFFFF) << 16),             // dim0 hi / dim1 lo
                   (int)(HD << 16),                        // dim1 hi / tile_dim0=64
                   32,                                     // tile_dim1=32
                   DD,                                     // dim0_stride lo32
                   0, 0 };
      i32x4 g2 = { 0, 0, 0, 0 };
      i32x4 g3 = { 0, 0, 0, 0 };
      i32x8 g4 = { 0, 0, 0, 0, 0, 0, 0, 0 };
      __builtin_amdgcn_tensor_load_to_lds(g0, g1, g2, g3, g4, 0);
      __builtin_amdgcn_s_wait_tensorcnt(0);
    }

    // ---- all threads: stage V tile transposed into LDS (Vt[d][j]) ----
    {
      const int j  = tid >> 3;          // 0..31
      const int c0 = (tid & 7) * 8;     // 0..56
      v8bf chunk = *(const v8bf*)(Vb + base + (size_t)(j0 + j) * DD + c0);
#pragma unroll
      for (int e = 0; e < 8; ++e) Vlds[(c0 + e) * 32 + j] = chunk[e];
    }
    __syncthreads();

    // ---- scores: 16x32 as two 16x16 WMMA tiles from shared K tile ----
    float s0[8], s1[8];
#pragma unroll
    for (int t = 0; t < 2; ++t) {
      const __bf16* kr = &Klds[(t * 16 + lcol) * 64];
      v16bf K0 = mk16(*(const v8bf*)(kr + half * 8),
                      *(const v8bf*)(kr + 16 + half * 8));
      v16bf K1 = mk16(*(const v8bf*)(kr + 32 + half * 8),
                      *(const v8bf*)(kr + 48 + half * 8));
      v8f c = {};
      c = wmma_bf16(Qa0, K0, c);
      c = wmma_bf16(Qa1, K1, c);
      const bool valid = kvmask[(size_t)b * SS + j0 + t * 16 + lcol] != 0;
      float* s = t ? s1 : s0;
#pragma unroll
      for (int r = 0; r < 8; ++r)
        s[r] = valid ? c[r] * 0.03125f : -3.0e38f;   // 1/sqrt(D) = 1/32
    }

    // ---- online softmax (row reductions across 16-lane halves) ----
    float rmax[8];
#pragma unroll
    for (int r = 0; r < 8; ++r) rmax[r] = fmaxf(s0[r], s1[r]);
#pragma unroll
    for (int off = 1; off < 16; off <<= 1)
#pragma unroll
      for (int r = 0; r < 8; ++r)
        rmax[r] = fmaxf(rmax[r], __shfl_xor(rmax[r], off, 32));

    float mnew[8], alpha[8], p0[8], p1[8], rsum[8];
#pragma unroll
    for (int r = 0; r < 8; ++r) {
      mnew[r]  = fmaxf(mrow[r], rmax[r]);
      alpha[r] = (mnew[r] <= -1.0e37f) ? 1.0f : __expf(mrow[r] - mnew[r]);
      p0[r] = (s0[r] <= -1.0e37f) ? 0.0f : __expf(s0[r] - mnew[r]);
      p1[r] = (s1[r] <= -1.0e37f) ? 0.0f : __expf(s1[r] - mnew[r]);
      rsum[r] = p0[r] + p1[r];
    }
#pragma unroll
    for (int off = 1; off < 16; off <<= 1)
#pragma unroll
      for (int r = 0; r < 8; ++r) rsum[r] += __shfl_xor(rsum[r], off, 32);
#pragma unroll
    for (int r = 0; r < 8; ++r) {
      lrow[r] = lrow[r] * alpha[r] + rsum[r];
      mrow[r] = mnew[r];
    }
#pragma unroll
    for (int g = 0; g < 4; ++g)
#pragma unroll
      for (int r = 0; r < 8; ++r) o[g][r] *= alpha[r];

    // ---- P -> per-wave LDS slice (same-wave LDS ops are in-order) ----
    __bf16* pw = &Plds[wv][0];
#pragma unroll
    for (int r = 0; r < 8; ++r) {
      pw[(half * 8 + r) * 32 + lcol]      = f2bf(p0[r]);
      pw[(half * 8 + r) * 32 + 16 + lcol] = f2bf(p1[r]);
    }

    // ---- O += P(16x32) x V(32x64): 1 A-frag + 4 B-frags, 4 WMMAs ----
    v16bf Pf = mk16(*(const v8bf*)(pw + lcol * 32 + half * 8),
                    *(const v8bf*)(pw + lcol * 32 + 16 + half * 8));
#pragma unroll
    for (int g = 0; g < 4; ++g) {
      const __bf16* vt = &Vlds[(g * 16 + lcol) * 32];
      v16bf Vf = mk16(*(const v8bf*)(vt + half * 8),
                      *(const v8bf*)(vt + 16 + half * 8));
      o[g] = wmma_bf16(Pf, Vf, o[g]);
    }
    __syncthreads();   // K/V tiles may be overwritten next iteration
  }

  // ---- normalize and store bf16 attention output ----
#pragma unroll
  for (int r = 0; r < 8; ++r) {
    float inv = 1.0f / lrow[r];
    __bf16* orow = Ob + base + (size_t)(q0 + half * 8 + r) * DD;
#pragma unroll
    for (int g = 0; g < 4; ++g)
      orow[g * 16 + lcol] = f2bf(o[g][r] * inv);
  }
}

// ---------------------------------------------------------------------------
// Host launch
// ---------------------------------------------------------------------------
extern "C" void kernel_launch(void* const* d_in, const int* in_sizes, int n_in,
                              void* d_out, int out_size, void* d_ws,
                              size_t ws_size, hipStream_t stream) {
  (void)in_sizes; (void)n_in; (void)out_size; (void)ws_size;

  const float* q              = (const float*)d_in[0];
  const unsigned char* kvmask = (const unsigned char*)d_in[1];
  const float* Wq             = (const float*)d_in[2];
  const float* Wk             = (const float*)d_in[3];
  const float* Wv             = (const float*)d_in[4];
  const float* Wo             = (const float*)d_in[5];
  const float* bo             = (const float*)d_in[6];

  char* ws = (char*)d_ws;
  const size_t ACT_B = (size_t)MM * DD * 2;   // 8 MiB per activation (bf16)
  const size_t W_B   = (size_t)DD * DD * 2;   // 2 MiB per weight (bf16)
  __bf16* qb  = (__bf16*)(ws);
  __bf16* Wqb = (__bf16*)(ws + ACT_B);
  __bf16* Wkb = (__bf16*)(ws + ACT_B + W_B);
  __bf16* Wvb = (__bf16*)(ws + ACT_B + 2 * W_B);
  __bf16* Wob = (__bf16*)(ws + ACT_B + 3 * W_B);
  __bf16* Qb  = (__bf16*)(ws + ACT_B + 4 * W_B);
  __bf16* Kb  = (__bf16*)(ws + 2 * ACT_B + 4 * W_B);
  __bf16* Vb  = (__bf16*)(ws + 3 * ACT_B + 4 * W_B);
  __bf16* Ob  = (__bf16*)(ws + 4 * ACT_B + 4 * W_B);

  const int nAct = MM * DD;       // 4194304
  const int nW   = DD * DD;       // 1048576
  cvt_f32_bf16<<<(nAct + 255) / 256, 256, 0, stream>>>(q,  qb,  nAct);
  cvt_f32_bf16<<<(nW   + 255) / 256, 256, 0, stream>>>(Wq, Wqb, nW);
  cvt_f32_bf16<<<(nW   + 255) / 256, 256, 0, stream>>>(Wk, Wkb, nW);
  cvt_f32_bf16<<<(nW   + 255) / 256, 256, 0, stream>>>(Wv, Wvb, nW);
  cvt_f32_bf16<<<(nW   + 255) / 256, 256, 0, stream>>>(Wo, Wob, nW);

  dim3 ggrid(DD / 256, MM / 64);  // (4, 64)
  gemm_bf16_wmma<<<ggrid, 256, 0, stream>>>(qb, Wqb, Qb, nullptr, nullptr,
                                            MM, DD, DD);
  gemm_bf16_wmma<<<ggrid, 256, 0, stream>>>(qb, Wkb, Kb, nullptr, nullptr,
                                            MM, DD, DD);
  gemm_bf16_wmma<<<ggrid, 256, 0, stream>>>(qb, Wvb, Vb, nullptr, nullptr,
                                            MM, DD, DD);

  attn_wmma<<<dim3(SS / 128, BB * HH), 256, 0, stream>>>(Qb, Kb, Vb, kvmask,
                                                         Ob);

  gemm_bf16_wmma<<<ggrid, 256, 0, stream>>>(Ob, Wob, nullptr, (float*)d_out,
                                            bo, MM, DD, DD);
}
